// RelationMessagePassing_65377992180480
// MI455X (gfx1250) — matrix operand order repaired and checked by reading, not compiled, not run on hardware
//
#include <hip/hip_runtime.h>
#include <hip/hip_bf16.h>

#define EDIM 64
#define SMOOTH 12.0f

typedef __attribute__((ext_vector_type(16))) __bf16 v16bf;
typedef __attribute__((ext_vector_type(2)))  __bf16 bf2;
typedef __attribute__((ext_vector_type(8)))  float  v8f;
typedef __attribute__((ext_vector_type(4)))  float  f4;

// Branch-free mish: x * tanh(softplus(x)) = x*(e^2+2e)/(e^2+2e+2), e=exp(x).
// Clamp keeps e^2 finite (x=30 -> e^2 ~ 1e26 < FLT_MAX); e underflow -> mish -> 0.
__device__ __forceinline__ float mishf(float x) {
    float e = __expf(fminf(x, 30.0f));     // v_exp_f32
    float n = __builtin_fmaf(e, e, 2.0f * e);
    return x * n * __builtin_amdgcn_rcpf(n + 2.0f);
}

// ---------------------------------------------------------------- weight cvt
__global__ void k_cvt_bf16(const float* __restrict__ src, __bf16* __restrict__ dst, int n) {
    int i = blockIdx.x * blockDim.x + threadIdx.x;
    if (i < n) dst[i] = (__bf16)src[i];
}

// ------------------------------------------------- relation residual MLP (WMMA)
// Block: MT 16-row M-tiles, D/16 waves (one 16-wide N-tile per wave).
// B operands (both layers' weights for this wave's N-tile) are register-cached
// once per block; the M-tile loop runs LDS->WMMA only.
template <int D, int ARITY, int MT>
__global__ __launch_bounds__(32 * (D / 16))
void k_relmsg(const float* __restrict__ emb,
              const int*   __restrict__ idx,
              const __bf16* __restrict__ Wi, const float* __restrict__ bi,
              const __bf16* __restrict__ Wo, const float* __restrict__ bo,
              float* __restrict__ msgs, int R)
{
    constexpr int NW   = D / 16;    // waves per block
    constexpr int KS   = D / 32;    // k-steps per GEMM
    constexpr int ROWS = MT * 16;   // relation rows per block

    __shared__ float  xf[ROWS][D];
    __shared__ __bf16 xb[ROWS][D];
    __shared__ __bf16 hb[ROWS][D];

    const int r0   = blockIdx.x * ROWS;
    const int tid  = threadIdx.x;
    const int nthr = 32 * NW;
    const int wave = tid >> 5;
    const int lane = tid & 31;
    const int ncol = wave * 16 + (lane & 15);
    const int mrow0 = (lane < 16) ? 0 : 8;   // C/D layout: VGPR r -> M=r (+8 hi half)
    const int hi8   = (lane < 16) ? 0 : 8;   // A layout: hi lanes hold K {8..15,24..31}
    const int bk0   = (lane < 16) ? 0 : 16;  // B layout: same K order as A (hi lanes K {8..15,24..31})
    const int arow  = lane & 15;

    // --- register-cache this wave's B-operand weight tiles (both layers)
    v16bf bwi[KS], bwo[KS];
    #pragma unroll
    for (int ks = 0; ks < KS; ++ks) {
        #pragma unroll
        for (int v = 0; v < 8; ++v) {
            int kk = ks * 32 + ((v < 4) ? 2 * v : 8 + 2 * v) + hi8;   // B[k][n] = W[n][k]
            bf2 p = *(const bf2*)&Wi[(size_t)ncol * D + kk];
            bwi[ks][2 * v] = p[0]; bwi[ks][2 * v + 1] = p[1];
            bf2 q = *(const bf2*)&Wo[(size_t)ncol * D + kk];
            bwo[ks][2 * v] = q[0]; bwo[ks][2 * v + 1] = q[1];
        }
    }
    const float bvi = bi[ncol];
    const float bvo = bo[ncol];

    // --- gather x = emb[idx] (ROWS x D); coalesced in 64-wide runs
    for (int i = tid; i < ROWS * D; i += nthr) {
        int row = i / D, col = i % D;
        int r = r0 + row; if (r >= R) r = R - 1;
        int o = idx[r * ARITY + col / EDIM];
        float v = emb[(size_t)o * EDIM + (col % EDIM)];
        xf[row][col] = v;
        xb[row][col] = (__bf16)v;
    }
    __syncthreads();

    for (int mt = 0; mt < MT; ++mt) {
        const int mbase = mt * 16;

        // GEMM1: h = mish(x @ Wi^T + bi)
        v8f acc;
        #pragma unroll
        for (int r = 0; r < 8; ++r) acc[r] = bvi;
        #pragma unroll
        for (int ks = 0; ks < KS; ++ks) {
            v16bf a;
            #pragma unroll
            for (int v = 0; v < 8; ++v) {
                int ka = ks * 32 + ((v < 4) ? 2 * v : 8 + 2 * v) + hi8;
                bf2 p = *(const bf2*)&xb[mbase + arow][ka];
                a[2 * v] = p[0]; a[2 * v + 1] = p[1];
            }
            acc = __builtin_amdgcn_wmma_f32_16x16x32_bf16(false, a, false, bwi[ks],
                                                          (short)0, acc, false, false);
        }
        #pragma unroll
        for (int r = 0; r < 8; ++r)
            hb[mbase + mrow0 + r][ncol] = (__bf16)mishf(acc[r]);
        __syncthreads();   // h-tile complete across all waves

        // GEMM2: m = x + h @ Wo^T + bo
        #pragma unroll
        for (int r = 0; r < 8; ++r) acc[r] = bvo;
        #pragma unroll
        for (int ks = 0; ks < KS; ++ks) {
            v16bf a;
            #pragma unroll
            for (int v = 0; v < 8; ++v) {
                int ka = ks * 32 + ((v < 4) ? 2 * v : 8 + 2 * v) + hi8;
                bf2 p = *(const bf2*)&hb[mbase + arow][ka];
                a[2 * v] = p[0]; a[2 * v + 1] = p[1];
            }
            acc = __builtin_amdgcn_wmma_f32_16x16x32_bf16(false, a, false, bwo[ks],
                                                          (short)0, acc, false, false);
        }
        // residual + write messages: msg row = rel*ARITY + ncol/64, elem = ncol%64
        #pragma unroll
        for (int r = 0; r < 8; ++r) {
            int m  = mbase + mrow0 + r;
            int rr = r0 + m;
            if (rr < R) {
                float val = xf[m][ncol] + acc[r];
                msgs[((size_t)rr * ARITY + ncol / EDIM) * EDIM + (ncol % EDIM)] = val;
            }
        }
        // next iteration writes a different hb slice -> no extra barrier needed
    }
}

// ---------------------------------------------------------------- aggregation
__global__ void k_agg_init(float* __restrict__ mx, float* __restrict__ s, int n) {
    for (int i = blockIdx.x * blockDim.x + threadIdx.x; i < n; i += gridDim.x * blockDim.x) {
        mx[i] = -__builtin_inff();
        s[i]  = 0.0f;
    }
}

// one float4 (global_load_b128) + one index load per 4 atomics
__global__ void k_scatter_max(const float* __restrict__ msgs, const int* __restrict__ idx,
                              float* __restrict__ mx, int nMsg) {
    int total = nMsg * (EDIM / 4);
    for (int i = blockIdx.x * blockDim.x + threadIdx.x; i < total; i += gridDim.x * blockDim.x) {
        int row = i >> 4, q = (i & 15) * 4;
        int seg = idx[row];
        f4 v = *(const f4*)&msgs[(size_t)row * EDIM + q];
        float* base = &mx[(size_t)seg * EDIM + q];
        #pragma unroll
        for (int j = 0; j < 4; ++j)
            __hip_atomic_fetch_max(&base[j], v[j], __ATOMIC_RELAXED, __HIP_MEMORY_SCOPE_AGENT);
    }
}

__global__ void k_fix_mx(float* __restrict__ mx, int n) {
    for (int i = blockIdx.x * blockDim.x + threadIdx.x; i < n; i += gridDim.x * blockDim.x) {
        float v = mx[i];
        if (__builtin_isinf(v) && v < 0.0f) mx[i] = 0.0f;
    }
}

__global__ void k_scatter_sum(const float* __restrict__ msgs, const int* __restrict__ idx,
                              const float* __restrict__ mx, float* __restrict__ s, int nMsg) {
    int total = nMsg * (EDIM / 4);
    for (int i = blockIdx.x * blockDim.x + threadIdx.x; i < total; i += gridDim.x * blockDim.x) {
        int row = i >> 4, q = (i & 15) * 4;
        int seg = idx[row];
        f4 v = *(const f4*)&msgs[(size_t)row * EDIM + q];
        const float* mbase = &mx[(size_t)seg * EDIM + q];
        float* sbase = &s[(size_t)seg * EDIM + q];
        #pragma unroll
        for (int j = 0; j < 4; ++j) {
            float e = __expf(SMOOTH * (v[j] - mbase[j]));   // v_exp_f32
            __hip_atomic_fetch_add(&sbase[j], e, __ATOMIC_RELAXED, __HIP_MEMORY_SCOPE_AGENT);
        }
    }
}

// ------------------------------------------------- update MLP (128 -> 128 -> 64)
__global__ __launch_bounds__(256)
void k_update(const float* __restrict__ emb,
              const float* __restrict__ mx, const float* __restrict__ ssum,
              const __bf16* __restrict__ Wi, const float* __restrict__ bi,
              const __bf16* __restrict__ Wo, const float* __restrict__ bo,
              float* __restrict__ out, int N)
{
    constexpr int D  = 128;
    constexpr int KS = D / 32;
    constexpr int MT = 4;
    constexpr int ROWS = MT * 16;
    __shared__ __bf16 ub[ROWS][D];
    __shared__ __bf16 hb[ROWS][D];

    const int r0  = blockIdx.x * ROWS;
    const int tid = threadIdx.x;
    const int wave = tid >> 5;
    const int lane = tid & 31;
    const int ncol = wave * 16 + (lane & 15);
    const int mrow0 = (lane < 16) ? 0 : 8;
    const int hi8   = (lane < 16) ? 0 : 8;
    const int bk0   = (lane < 16) ? 0 : 16;
    const int arow  = lane & 15;

    // register-cache weights: Wi for all 8 waves; Wo only for waves 0..3 (64 outs)
    v16bf bwi[KS], bwo[KS];
    #pragma unroll
    for (int ks = 0; ks < KS; ++ks) {
        #pragma unroll
        for (int v = 0; v < 8; ++v) {
            int kk = ks * 32 + ((v < 4) ? 2 * v : 8 + 2 * v) + hi8;
            bf2 p = *(const bf2*)&Wi[(size_t)ncol * D + kk];
            bwi[ks][2 * v] = p[0]; bwi[ks][2 * v + 1] = p[1];
        }
    }
    float bvi = bi[ncol];
    float bvo = 0.0f;
    if (wave < 4) {
        #pragma unroll
        for (int ks = 0; ks < KS; ++ks) {
            #pragma unroll
            for (int v = 0; v < 8; ++v) {
                int kk = ks * 32 + ((v < 4) ? 2 * v : 8 + 2 * v) + hi8;
                bf2 q = *(const bf2*)&Wo[(size_t)ncol * D + kk];
                bwo[ks][2 * v] = q[0]; bwo[ks][2 * v + 1] = q[1];
            }
        }
        bvo = bo[ncol];
    }

    // gather u = [logsumexp_msg, emb]; final LSE math fused here
    for (int i = tid; i < ROWS * D; i += 256) {
        int row = i / D, col = i % D;
        int o = r0 + row; if (o >= N) o = N - 1;
        float v;
        if (col < EDIM) {
            size_t off = (size_t)o * EDIM + col;
            v = __logf(ssum[off] + 1e-16f) * (1.0f / SMOOTH) + mx[off];   // v_log_f32
        } else {
            v = emb[(size_t)o * EDIM + (col - EDIM)];
        }
        ub[row][col] = (__bf16)v;
    }
    __syncthreads();

    for (int mt = 0; mt < MT; ++mt) {
        const int mbase = mt * 16;
        v8f acc;
        #pragma unroll
        for (int r = 0; r < 8; ++r) acc[r] = bvi;
        #pragma unroll
        for (int ks = 0; ks < KS; ++ks) {
            v16bf a;
            #pragma unroll
            for (int v = 0; v < 8; ++v) {
                int ka = ks * 32 + ((v < 4) ? 2 * v : 8 + 2 * v) + hi8;
                bf2 p = *(const bf2*)&ub[mbase + arow][ka];
                a[2 * v] = p[0]; a[2 * v + 1] = p[1];
            }
            acc = __builtin_amdgcn_wmma_f32_16x16x32_bf16(false, a, false, bwi[ks],
                                                          (short)0, acc, false, false);
        }
        #pragma unroll
        for (int r = 0; r < 8; ++r)
            hb[mbase + mrow0 + r][ncol] = (__bf16)mishf(acc[r]);
        __syncthreads();

        if (wave < 4) {   // second layer: 64 output columns
            #pragma unroll
            for (int r = 0; r < 8; ++r) acc[r] = bvo;
            #pragma unroll
            for (int ks = 0; ks < KS; ++ks) {
                v16bf a;
                #pragma unroll
                for (int v = 0; v < 8; ++v) {
                    int ka = ks * 32 + ((v < 4) ? 2 * v : 8 + 2 * v) + hi8;
                    bf2 p = *(const bf2*)&hb[mbase + arow][ka];
                    a[2 * v] = p[0]; a[2 * v + 1] = p[1];
                }
                acc = __builtin_amdgcn_wmma_f32_16x16x32_bf16(false, a, false, bwo[ks],
                                                              (short)0, acc, false, false);
            }
            #pragma unroll
            for (int r = 0; r < 8; ++r) {
                int m = mbase + mrow0 + r;
                int o = r0 + m;
                if (o < N)
                    out[(size_t)o * EDIM + ncol] = emb[(size_t)o * EDIM + ncol] + acc[r];
            }
        }
    }
}

// ---------------------------------------------------------------- launch
extern "C" void kernel_launch(void* const* d_in, const int* in_sizes, int n_in,
                              void* d_out, int out_size, void* d_ws, size_t ws_size,
                              hipStream_t stream)
{
    const float* emb  = (const float*)d_in[0];
    const int*   idxU = (const int*)d_in[1];
    const int*   idxB = (const int*)d_in[2];
    const int*   idxT = (const int*)d_in[3];
    const float* uWi = (const float*)d_in[4];  const float* uBi = (const float*)d_in[5];
    const float* uWo = (const float*)d_in[6];  const float* uBo = (const float*)d_in[7];
    const float* bWi = (const float*)d_in[8];  const float* bBi = (const float*)d_in[9];
    const float* bWo = (const float*)d_in[10]; const float* bBo = (const float*)d_in[11];
    const float* tWi = (const float*)d_in[12]; const float* tBi = (const float*)d_in[13];
    const float* tWo = (const float*)d_in[14]; const float* tBo = (const float*)d_in[15];
    const float* pWi = (const float*)d_in[16]; const float* pBi = (const float*)d_in[17];
    const float* pWo = (const float*)d_in[18]; const float* pBo = (const float*)d_in[19];

    const int nU = in_sizes[1];          // 200000 messages (arity 1)
    const int nB = in_sizes[2];          // 600000 messages
    const int nT = in_sizes[3];          // 300000 messages
    const int RU = nU, RB = nB / 2, RT = nT / 3;
    const int NOBJ = in_sizes[0] / EDIM; // 100000

    char* ws = (char*)d_ws;
    float* msgU = (float*)ws; ws += (size_t)nU * EDIM * sizeof(float);
    float* msgB = (float*)ws; ws += (size_t)nB * EDIM * sizeof(float);
    float* msgT = (float*)ws; ws += (size_t)nT * EDIM * sizeof(float);
    float* mx   = (float*)ws; ws += (size_t)NOBJ * EDIM * sizeof(float);
    float* ssum = (float*)ws; ws += (size_t)NOBJ * EDIM * sizeof(float);
    __bf16* uWiB = (__bf16*)ws; ws += 64 * 64 * sizeof(__bf16);
    __bf16* uWoB = (__bf16*)ws; ws += 64 * 64 * sizeof(__bf16);
    __bf16* bWiB = (__bf16*)ws; ws += 128 * 128 * sizeof(__bf16);
    __bf16* bWoB = (__bf16*)ws; ws += 128 * 128 * sizeof(__bf16);
    __bf16* tWiB = (__bf16*)ws; ws += 192 * 192 * sizeof(__bf16);
    __bf16* tWoB = (__bf16*)ws; ws += 192 * 192 * sizeof(__bf16);
    __bf16* pWiB = (__bf16*)ws; ws += 128 * 128 * sizeof(__bf16);
    __bf16* pWoB = (__bf16*)ws; ws += 64 * 128 * sizeof(__bf16);

    auto cvt = [&](const float* s, __bf16* d, int n) {
        k_cvt_bf16<<<(n + 255) / 256, 256, 0, stream>>>(s, d, n);
    };
    cvt(uWi, uWiB, 64 * 64);   cvt(uWo, uWoB, 64 * 64);
    cvt(bWi, bWiB, 128 * 128); cvt(bWo, bWoB, 128 * 128);
    cvt(tWi, tWiB, 192 * 192); cvt(tWo, tWoB, 192 * 192);
    cvt(pWi, pWiB, 128 * 128); cvt(pWo, pWoB, 64 * 128);

    // per-relation residual MLPs -> message buffers
    // ROWS per block chosen so static LDS stays <= 64 KB: 64/48/32 rows.
    k_relmsg<64, 1, 4><<<(RU + 63) / 64, 128, 0, stream>>>(emb, idxU, uWiB, uBi, uWoB, uBo, msgU, RU);
    k_relmsg<128, 2, 3><<<(RB + 47) / 48, 256, 0, stream>>>(emb, idxB, bWiB, bBi, bWoB, bBo, msgB, RB);
    k_relmsg<192, 3, 2><<<(RT + 31) / 32, 384, 0, stream>>>(emb, idxT, tWiB, tBi, tWoB, tBo, msgT, RT);

    // smooth log-sum-exp aggregation
    const int nSeg = NOBJ * EDIM;
    k_agg_init<<<4096, 256, 0, stream>>>(mx, ssum, nSeg);
    k_scatter_max<<<8192, 256, 0, stream>>>(msgU, idxU, mx, nU);
    k_scatter_max<<<8192, 256, 0, stream>>>(msgB, idxB, mx, nB);
    k_scatter_max<<<8192, 256, 0, stream>>>(msgT, idxT, mx, nT);
    k_fix_mx<<<4096, 256, 0, stream>>>(mx, nSeg);
    k_scatter_sum<<<8192, 256, 0, stream>>>(msgU, idxU, mx, ssum, nU);
    k_scatter_sum<<<8192, 256, 0, stream>>>(msgB, idxB, mx, ssum, nB);
    k_scatter_sum<<<8192, 256, 0, stream>>>(msgT, idxT, mx, ssum, nT);

    // update MLP + residual -> d_out
    k_update<<<(NOBJ + 63) / 64, 256, 0, stream>>>(emb, mx, ssum, pWiB, pBi, pWoB, pBo,
                                                   (float*)d_out, NOBJ);
}
